// TopK_custom4_train_tau_31258771981053
// MI455X (gfx1250) — matrix operand order repaired and checked by reading, not compiled, not run on hardware
//
#include <hip/hip_runtime.h>
#include <math.h>

// Problem constants (from reference): BS=256, N=2048, K=64, eps=0.1, 200 iters
#define BSZ 256
#define NN 2048
#define KK 64
#define KC 65          // K+1 columns
#define GSTRIDE 72     // f16 columns per G row, padded (tail zeros), 144 B = 16B-aligned stride
#define MAX_ITER 200
#define SMALLV 1e-20f
#define MUV (1.0f/2048.0f)
#define LOG2E 1.4426950408889634f

typedef _Float16 h2  __attribute__((ext_vector_type(2)));
typedef _Float16 h8  __attribute__((ext_vector_type(8)));
typedef _Float16 v16h __attribute__((ext_vector_type(16)));
typedef float    v8f  __attribute__((ext_vector_type(8)));

// ---- workspace layout (float/uint words) ----
#define WS_MINENC 0
#define WS_MAXENC 1
#define WS_NEGCNT 2
#define WS_FILL   3
#define WS_C2     4                     // -log2(e)/(eps*Cmax)
#define WS_TAU    16                    // [BSZ]
#define WS_TOPK   (WS_TAU + BSZ)        // [BSZ][64]
#define WS_CS0    (WS_TOPK + BSZ*KK)    // [BSZ][64]  Gamma0 column sums
#define WS_U      (WS_CS0 + BSZ*KK)     // [BSZ][NN]
#define WS_V      (WS_U + BSZ*NN)       // [BSZ][KC]
#define WS_PART   (WS_V + BSZ*KC)       // [BSZ] norm partials

// ---- dynamic LDS layout for the Sinkhorn kernel (312,160 B < 320 KB/WGP) ----
#define L_GH   0                        // f16 G[2049][72] = 295,056 B (row 2048 = zero pad)
#define L_U    295056                   // f32 u[2048]     = 8,192 B
#define L_CP   303248                   // f32 colpart[32][66] = 8,448 B
#define L_VF   311696                   // f32 v[68]       = 272 B
#define L_VH   311968                   // f16 vh[96]      = 192 B (65..95 stay zero -> K padding)
#define LDS_BYTES 312160

// order-preserving float<->uint encode for integer atomic min/max (deterministic)
__device__ __forceinline__ unsigned encf(float f){
  unsigned u = __float_as_uint(f);
  return (u & 0x80000000u) ? ~u : (u | 0x80000000u);
}
__device__ __forceinline__ float decf(unsigned e){
  unsigned u = (e & 0x80000000u) ? (e & 0x7fffffffu) : ~e;
  return __uint_as_float(u);
}

__global__ void k_init(unsigned* wsw){
  wsw[WS_MINENC] = 0xFFFFFFFFu;
  wsw[WS_MAXENC] = 0u;
  wsw[WS_NEGCNT] = 0u;
}

__global__ void __launch_bounds__(1024) k_minmax(const float* __restrict__ sc, unsigned* wsw){
  __shared__ unsigned rmin[32], rmax[32];
  int gid = blockIdx.x * 1024 + threadIdx.x;
  float f = sc[gid];
  bool neg = (f == -INFINITY);
  if (neg) atomicAdd(&wsw[WS_NEGCNT], 1u);
  unsigned e = encf(neg ? INFINITY : f);   // min ignores -inf entries
  unsigned E = encf(f);                    // max keeps them (can't win)
  #pragma unroll
  for (int o = 16; o; o >>= 1){
    unsigned eo = (unsigned)__shfl_xor((int)e, o, 32);
    unsigned Eo = (unsigned)__shfl_xor((int)E, o, 32);
    e = e < eo ? e : eo;  E = E > Eo ? E : Eo;
  }
  int wv = threadIdx.x >> 5, lane = threadIdx.x & 31;
  if (lane == 0){ rmin[wv] = e; rmax[wv] = E; }
  __syncthreads();
  if (threadIdx.x == 0){
    unsigned m = rmin[0], M = rmax[0];
    for (int z = 1; z < 32; ++z){ m = m < rmin[z] ? m : rmin[z]; M = M > rmax[z] ? M : rmax[z]; }
    atomicMin(&wsw[WS_MINENC], m);
    atomicMax(&wsw[WS_MAXENC], M);
  }
}

__global__ void k_scalars(unsigned* wsw, float* wsf){
  float smin = decf(wsw[WS_MINENC]);
  float smax = decf(wsw[WS_MAXENC]);
  float fill = smin - (smax - smin);
  float lo = (wsw[WS_NEGCNT] > 0u) ? fill : smin;      // min of filled s
  // C = (s-a)^2 / Cmax, a in [0,64]; max of (s-a)^2 is at the corners
  float m1 = fmaxf(lo*lo, (lo-64.0f)*(lo-64.0f));
  float m2 = fmaxf(smax*smax, (smax-64.0f)*(smax-64.0f));
  float cmax = fmaxf(m1, m2);
  wsf[WS_FILL] = fill;
  wsf[WS_C2]   = -LOG2E / (0.1f * cmax);   // G = exp2(d*d*c2)
}

// per-batch: bitonic sort in LDS -> tau, topk; then Gamma0 column sums
__global__ void __launch_bounds__(1024) k_sort_tau(const float* __restrict__ sc,
                                                   const float* __restrict__ W,
                                                   float* wsf){
  __shared__ float sa[NN];
  __shared__ float red[32];
  __shared__ float tk[KK];
  __shared__ float tauS;
  __shared__ float part[KK][16];
  int b = blockIdx.x, tid = threadIdx.x;
  float fill = wsf[WS_FILL];
  for (int i = tid; i < NN; i += 1024){
    float s = sc[b*NN + i];
    if (s == -INFINITY) s = fill;
    sa[i] = s;
  }
  __syncthreads();
  // bitonic sort ascending, 2048 elements
  for (int k = 2; k <= NN; k <<= 1){
    for (int j = k >> 1; j > 0; j >>= 1){
      for (int t = tid; t < NN; t += 1024){
        int ixj = t ^ j;
        if (ixj > t){
          float a = sa[t], c = sa[ixj];
          bool up = ((t & k) == 0);
          if ((a > c) == up){ sa[t] = c; sa[ixj] = a; }
        }
      }
      __syncthreads();
    }
  }
  // tau = sum(sorted * W)  (fixed-order tree -> deterministic)
  float p = sa[tid]*W[tid] + sa[tid+1024]*W[tid+1024];
  #pragma unroll
  for (int o = 16; o; o >>= 1) p += __shfl_xor(p, o, 32);
  int wv = tid >> 5, lane = tid & 31;
  if (lane == 0) red[wv] = p;
  __syncthreads();
  if (tid == 0){
    float t = 0.f;
    for (int z = 0; z < 32; ++z) t += red[z];
    tauS = t; wsf[WS_TAU + b] = t;
  }
  if (tid < KK){ float v = sa[NN-1-tid]; tk[tid] = v; wsf[WS_TOPK + b*KK + tid] = v; }
  __syncthreads();
  // Gamma0 column sums: 16 threads per column j
  int j = tid >> 4, sub = tid & 15;
  float itau = LOG2E / tauS;
  float tj = tk[j];
  float cs = 0.f;
  for (int i = sub; i < NN; i += 16){
    float s = sc[b*NN + i];
    if (s == -INFINITY) s = fill;
    float q = __builtin_exp2f(fabsf(tj - s) * itau);       // exp(|d|/tau)
    cs += __builtin_amdgcn_rcpf(1.0f + q) + SMALLV;        // sigmoid(-|d|/tau) + SMALL
  }
  part[j][sub] = cs;
  __syncthreads();
  if (tid < KK){
    float t = 0.f;
    #pragma unroll
    for (int z = 0; z < 16; ++z) t += part[tid][z];
    wsf[WS_CS0 + b*KK + tid] = t;
  }
}

// The hot kernel: one workgroup per batch; G (f16) lives in LDS for all 200 iterations.
__global__ void __launch_bounds__(1024) k_sinkhorn(const float* __restrict__ sc,
                                                   const float* __restrict__ wsf,
                                                   float* __restrict__ uo,
                                                   float* __restrict__ vo){
  extern __shared__ char lds[];
  _Float16* Gh = (_Float16*)(lds + L_GH);
  float*    uL = (float*)(lds + L_U);
  float*    cp = (float*)(lds + L_CP);
  float*    vf = (float*)(lds + L_VF);
  _Float16* vh = (_Float16*)(lds + L_VH);
  int b = blockIdx.x, tid = threadIdx.x;
  int wv = tid >> 5, lane = tid & 31, half = lane >> 4;
  float fill = wsf[WS_FILL], c2 = wsf[WS_C2];

  // ---- build G[i][k] = exp(-(s_i - (64-k))^2 / (eps*Cmax)) as f16, tail zeros ----
  for (int i = tid; i < NN; i += 1024){
    float s = sc[b*NN + i];
    if (s == -INFINITY) s = fill;
    _Float16* row = Gh + i*GSTRIDE;
    float d0 = s - 64.0f;                 // d0 + k = s - (64-k)
    #pragma unroll 1
    for (int c = 0; c < 8; ++c){
      h8 g;
      #pragma unroll
      for (int q = 0; q < 8; ++q){
        float d = d0 + (float)(c*8 + q);
        g[q] = (_Float16)__builtin_exp2f(d*d*c2);
      }
      *(h8*)(row + c*8) = g;
    }
    { h8 g = {};
      float d = d0 + 64.0f;               // k = 64 (anchor 0)
      g[0] = (_Float16)__builtin_exp2f(d*d*c2);
      *(h8*)(row + 64) = g; }             // k=65..71 zeroed
  }
  if (tid < 9){ h8 z = {}; *(h8*)(Gh + NN*GSTRIDE + tid*8) = z; }  // pad row
  if (tid < 96) vh[tid] = (tid < KC) ? (_Float16)(1.0f/65.0f) : (_Float16)0.0f;
  if (tid < 68) vf[tid] = 1.0f/65.0f;
  __syncthreads();

  #pragma unroll 1
  for (int it = 0; it < MAX_ITER; ++it){
    // ---------- row phase: u_i = mu / sum_k G[i,k] v_k  (WMMA f16, K padded to 96) ----------
    // B = broadcast of v: lane-half h loads K = c*32 + h*8 + 0..7 and +16 (matches A) -> B[k,n]=v_k.
    v16h bf[3];
    #pragma unroll
    for (int c = 0; c < 3; ++c){
      const _Float16* vp = vh + c*32 + half*8;
      h8 blo = *(const h8*)vp;
      h8 bhi = *(const h8*)(vp + 16);
      #pragma unroll
      for (int q = 0; q < 8; ++q){ bf[c][q] = blo[q]; bf[c][8+q] = bhi[q]; }
    }
    #pragma unroll 1
    for (int t = 0; t < 4; ++t){
      int m0 = (wv*4 + t) * 16;
      int row = m0 + (lane & 15);
      // Issue all six b128 LDS loads first (one wait point), then the three
      // chained WMMAs back-to-back so the matrix pipe runs without per-chunk stalls.
      v16h a[3];
      #pragma unroll
      for (int c = 0; c < 3; ++c){
        // A layout: dwords0-3 <- K = c*32 + half*8 + 0..7 ; dwords4-7 <- +16
        const _Float16* gp = Gh + row*GSTRIDE + c*32 + half*8;
        h8 alo = *(const h8*)gp;
        h8 ahi = *(const h8*)(gp + 16);
        #pragma unroll
        for (int q = 0; q < 8; ++q){ a[c][q] = alo[q]; a[c][8+q] = ahi[q]; }
      }
      v8f acc = {};
      acc = __builtin_amdgcn_wmma_f32_16x16x32_f16(false, a[0], false, bf[0], (short)0, acc, false, false);
      acc = __builtin_amdgcn_wmma_f32_16x16x32_f16(false, a[1], false, bf[1], (short)0, acc, false, false);
      acc = __builtin_amdgcn_wmma_f32_16x16x32_f16(false, a[2], false, bf[2], (short)0, acc, false, false);
      // D: VGPR r holds rows m0+r (lanes 0-15) / m0+8+r (lanes 16-31); all N columns equal.
      if ((lane & 15) == 0){
        int rb = m0 + half*8;
        #pragma unroll
        for (int r = 0; r < 8; ++r) uL[rb + r] = MUV * __builtin_amdgcn_rcpf(acc[r]);
      }
    }
    __syncthreads();
    // ---------- column phase: v_k = nu_k / sum_i G[i,k] u_i ----------
    {
      float a0 = 0.f, a1 = 0.f;           // lane owns columns 2*lane, 2*lane+1 (bank-conflict-free)
      int base = wv * 64;
      #pragma unroll 1
      for (int ii = 0; ii < 64; ii += 4){
        float4 uq = *(const float4*)&uL[base + ii];
        #pragma unroll
        for (int q = 0; q < 4; ++q){
          h2 g = *(const h2*)(Gh + (base + ii + q)*GSTRIDE + 2*lane);
          float uu = ((const float*)&uq)[q];
          a0 += (float)g[0] * uu;
          a1 += (float)g[1] * uu;
        }
      }
      *(float2*)&cp[wv*66 + 2*lane] = make_float2(a0, a1);
      // column 64: butterfly over the wave
      int r0 = base + lane;
      float cl = (float)Gh[r0*GSTRIDE + 64]      * uL[r0]
               + (float)Gh[(r0+32)*GSTRIDE + 64] * uL[r0+32];
      #pragma unroll
      for (int o = 16; o; o >>= 1) cl += __shfl_xor(cl, o, 32);
      if (lane == 0) cp[wv*66 + 64] = cl;
    }
    __syncthreads();
    if (tid < KC){
      float s2 = 0.f;
      #pragma unroll 1
      for (int w2 = 0; w2 < 32; ++w2) s2 += cp[w2*66 + tid];
      float nu = (tid < KK) ? MUV : (1984.0f/2048.0f);
      float v = nu * __builtin_amdgcn_rcpf(s2);
      vf[tid] = v; vh[tid] = (_Float16)v;
    }
    __syncthreads();
  }
  for (int i = tid; i < NN; i += 1024) uo[b*NN + i] = uL[i];
  if (tid < KC) vo[b*KC + tid] = vf[tid];
}

// Final pass: A = u*G*v*n (exact f32 G), Gamma0 on the fly, norm partial per batch.
__global__ void __launch_bounds__(1024) k_output(const float* __restrict__ sc,
                                                 const float* __restrict__ wsf,
                                                 float* __restrict__ out,
                                                 float* __restrict__ part){
  __shared__ float tk[KK], ics[KK], vsh[KC], red[32];
  int b = blockIdx.x, tid = threadIdx.x;
  if (tid < KK){
    tk[tid]  = wsf[WS_TOPK + b*KK + tid];
    ics[tid] = 1.0f / (wsf[WS_CS0 + b*KK + tid] * 2048.0f);  // /colsum /n
  }
  if (tid < KC) vsh[tid] = wsf[WS_V + b*KC + tid];
  __syncthreads();
  float fill = wsf[WS_FILL], c2 = wsf[WS_C2];
  float itau = LOG2E / wsf[WS_TAU + b];
  float sq = 0.f;
  for (int i = tid; i < NN; i += 1024){
    float s = sc[b*NN + i];
    if (s == -INFINITY) s = fill;
    float u = wsf[WS_U + b*NN + i];
    float rs0 = 0.f;
    float* op = out + ((size_t)(b*NN + i)) * KK;
    #pragma unroll 1
    for (int j = 0; j < KK; j += 4){
      float4 o4;
      #pragma unroll
      for (int q = 0; q < 4; ++q){
        int jj = j + q;
        float d   = s - (float)(64 - jj);
        float gam = u * __builtin_exp2f(d*d*c2) * vsh[jj];
        float qq  = __builtin_exp2f(fabsf(tk[jj] - s) * itau);
        float g0  = (__builtin_amdgcn_rcpf(1.0f + qq) + SMALLV) * ics[jj];
        rs0 += g0;
        float df = gam - g0; sq += df*df;
        ((float*)&o4)[q] = gam * 2048.0f;
      }
      *(float4*)(op + j) = o4;
    }
    float l0 = fminf(fmaxf(MUV - rs0, SMALLV), 1.0f - SMALLV);
    float gl = u * __builtin_exp2f(s*s*c2) * vsh[KK];   // column 64, anchor 0
    float df = gl - l0; sq += df*df;
  }
  #pragma unroll
  for (int o = 16; o; o >>= 1) sq += __shfl_xor(sq, o, 32);
  int wv = tid >> 5, lane = tid & 31;
  if (lane == 0) red[wv] = sq;
  __syncthreads();
  if (tid == 0){
    float t = 0.f;
    for (int z = 0; z < 32; ++z) t += red[z];
    part[b] = t;
  }
}

__global__ void k_norm(const float* __restrict__ part, float* __restrict__ out){
  if (threadIdx.x == 0){
    float t = 0.f;
    for (int i = 0; i < BSZ; ++i) t += part[i];   // fixed order -> deterministic
    out[(size_t)BSZ * NN * KK] = sqrtf(t);
  }
}

extern "C" void kernel_launch(void* const* d_in, const int* in_sizes, int n_in,
                              void* d_out, int out_size, void* d_ws, size_t ws_size,
                              hipStream_t stream){
  const float* sc = (const float*)d_in[0];   // scores (256, 2048) f32
  const float* W  = (const float*)d_in[1];   // W (2048,) f32
  float*    out = (float*)d_out;
  float*    wsf = (float*)d_ws;
  unsigned* wsw = (unsigned*)d_ws;

  // allow >64KB dynamic LDS for the Sinkhorn kernel (idempotent; ignore rc)
  (void)hipFuncSetAttribute(reinterpret_cast<const void*>(k_sinkhorn),
                            hipFuncAttributeMaxDynamicSharedMemorySize, LDS_BYTES);

  k_init    <<<1, 1, 0, stream>>>(wsw);
  k_minmax  <<<(BSZ*NN)/1024, 1024, 0, stream>>>(sc, wsw);
  k_scalars <<<1, 1, 0, stream>>>(wsw, wsf);
  k_sort_tau<<<BSZ, 1024, 0, stream>>>(sc, W, wsf);
  k_sinkhorn<<<BSZ, 1024, LDS_BYTES, stream>>>(sc, wsf, wsf + WS_U, wsf + WS_V);
  k_output  <<<BSZ, 1024, 0, stream>>>(sc, wsf, out, wsf + WS_PART);
  k_norm    <<<1, 1, 0, stream>>>(wsf + WS_PART, out);
}
